// MinGRU_36361193128562
// MI455X (gfx1250) — matrix operand order repaired and checked by reading, not compiled, not run on hardware
//
#include <hip/hip_runtime.h>

// Problem constants (fixed by the reference: B=8, T=4096, D_in=H=D_out=1024)
#define BATCH   8
#define TSEQ    4096
#define TOTAL_M 32768      // B*T
#define DIN     1024       // K for both GEMMs
#define HID     1024
#define DOUT    1024
#define NCHUNK  16         // scan chunks per sequence
#define CLEN    256        // T / NCHUNK

typedef __attribute__((ext_vector_type(16))) __bf16 v16bf;
typedef __attribute__((ext_vector_type(8)))  float  v8f;

union Frag32 { uint4 q[2]; v16bf v; };            // 32 bytes = 8 VGPRs per lane
union Pack8  { unsigned short s[8]; uint4 q; };   // 8 bf16 packed for b128 stores

// ---------- helpers ----------

__device__ __forceinline__ unsigned short f32_to_bf16_rne(float x) {
  unsigned u = __float_as_uint(x);
  unsigned r = 0x7FFFu + ((u >> 16) & 1u);
  return (unsigned short)((u + r) >> 16);
}

// split f32 into bf16 hi (truncated, exact) + bf16 lo (rounded residual)
__device__ __forceinline__ void split_hi_lo(float x, unsigned short* hi, unsigned short* lo) {
  unsigned uh = __float_as_uint(x) & 0xFFFF0000u;
  *hi = (unsigned short)(uh >> 16);
  *lo = f32_to_bf16_rne(x - __uint_as_float(uh));
}

__device__ __forceinline__ float softplusf(float x) {
  return fmaxf(x, 0.0f) + log1pf(expf(-fabsf(x)));
}

__device__ __forceinline__ float logaddexpf_(float p, float q) {
  float m = fmaxf(p, q);
  if (!(m > -INFINITY)) return m;              // both -inf
  return m + log1pf(expf(-fabsf(p - q)));
}

// CDNA5 async copy: one 16B chunk per lane, global -> LDS, tracked by ASYNCcnt.
__device__ __forceinline__ void async_copy16(const void* lds_dst, const void* gsrc) {
  unsigned lds = (unsigned)(uintptr_t)lds_dst;   // flat addr low 32 bits == LDS offset
  unsigned long long g = (unsigned long long)(uintptr_t)gsrc;
  asm volatile("global_load_async_to_lds_b128 %0, %1, off"
               :: "v"(lds), "v"(g) : "memory");
}
__device__ __forceinline__ void wait_async0() {
  asm volatile("s_wait_asynccnt 0x0" ::: "memory");
}

// ---------- pre-conversion kernels ----------

// x (f32, row-major [M][K]) -> bf16 hi/lo planes, 8 elements/thread, all b128.
__global__ __launch_bounds__(256)
void convert_x(const float* __restrict__ in,
               unsigned short* __restrict__ hi, unsigned short* __restrict__ lo) {
  const long e = ((long)blockIdx.x * 256 + threadIdx.x) * 8;
  float4 f0 = *(const float4*)(in + e);
  float4 f1 = *(const float4*)(in + e + 4);
  float v[8] = {f0.x, f0.y, f0.z, f0.w, f1.x, f1.y, f1.z, f1.w};
  Pack8 ph, pl;
#pragma unroll
  for (int j = 0; j < 8; ++j) split_hi_lo(v[j], &ph.s[j], &pl.s[j]);
  *(uint4*)(hi + e) = ph.q;
  *(uint4*)(lo + e) = pl.q;
}

// W (f32, [K][N]) -> transposed bf16 hi/lo planes [N][K], K = 1024.
// One-time cost on tiny matrices; makes B-fragments contiguous in K.
__global__ __launch_bounds__(256)
void convert_wT(const float* __restrict__ in, int N,
                unsigned short* __restrict__ hi, unsigned short* __restrict__ lo) {
  const int t  = blockIdx.x * 256 + threadIdx.x;   // N*K/8 threads
  const int n  = t >> 7;                           // K/8 = 128 chunks per row
  const int kq = (t & 127) * 8;
  Pack8 ph, pl;
#pragma unroll
  for (int j = 0; j < 8; ++j) {
    float v = in[(long)(kq + j) * N + n];
    split_hi_lo(v, &ph.s[j], &pl.s[j]);
  }
  *(uint4*)(hi + (long)n * DIN + kq) = ph.q;
  *(uint4*)(lo + (long)n * DIN + kq) = pl.q;
}

// ---------- split-bf16 WMMA GEMM (bf16 operands, async double-buffered LDS) ----------
// Workgroup tile: 64(M) x 64(N), 8 waves = 4(M) x 2(N); each wave: two 16x16
// accumulators per B-tile, K-tile = 32.
// A planes: bf16 [M][1024] hi/lo.  B planes: bf16 transposed [N][1024] hi/lo.
// MODE 0: N spans the hidden half; the matching gate tile (row n+1024 of the
//         transposed W_hg) is computed simultaneously and the softplus/log_g
//         epilogue is fused -> writes log_coeffs (out0), log_values (out1).
// MODE 1: plain f32 store to out0.

#define LDK 40   // padded LDS row (ushorts): 80B stride, 16B aligned, conflict-free

template <int MODE>
__global__ __launch_bounds__(256)
void wmma_gemm(const unsigned short* __restrict__ Ahi,
               const unsigned short* __restrict__ Alo,
               const unsigned short* __restrict__ Bhi,
               const unsigned short* __restrict__ Blo,
               float* __restrict__ out0,
               float* __restrict__ out1) {
  constexpr int NB = (MODE == 0) ? 2 : 1;

  __shared__ __align__(16) unsigned short sAhi[2][64][LDK];
  __shared__ __align__(16) unsigned short sAlo[2][64][LDK];
  __shared__ __align__(16) unsigned short sBhi[2][NB][64][LDK];  // [n][k] tiles
  __shared__ __align__(16) unsigned short sBlo[2][NB][64][LDK];

  const int tid  = threadIdx.x;
  const int lane = tid & 31;
  const int wave = tid >> 5;
  const int l16  = lane & 15;
  const bool top = lane >= 16;
  const int mw = (wave & 3) * 16;   // wave's M offset in tile
  const int nw = (wave >> 2) * 32;  // wave's N offset in tile

  const long m0 = (long)blockIdx.y * 64;
  const int  n0 = blockIdx.x * 64;

  // async staging assignment: 64 rows x 32 k bf16 = 4KB/plane; 16B per thread
  const int srow = tid >> 2;          // 0..63
  const int sseg = (tid & 3) * 8;     // 0,8,16,24 (ushorts)

  v8f acc[NB][2];
#pragma unroll
  for (int tb = 0; tb < NB; ++tb)
#pragma unroll
    for (int s = 0; s < 2; ++s) acc[tb][s] = (v8f)(0.0f);

  const long aOff = (m0 + srow) * (long)DIN + sseg;

  auto stage_all = [&](int buf, int k0) {
    async_copy16(&sAhi[buf][srow][sseg], Ahi + aOff + k0);
    async_copy16(&sAlo[buf][srow][sseg], Alo + aOff + k0);
#pragma unroll
    for (int tb = 0; tb < NB; ++tb) {
      const long bOff = (long)(n0 + tb * HID + srow) * DIN + sseg + k0;
      async_copy16(&sBhi[buf][tb][srow][sseg], Bhi + bOff);
      async_copy16(&sBlo[buf][tb][srow][sseg], Blo + bOff);
    }
  };

  stage_all(0, 0);
  wait_async0();
  __syncthreads();

  for (int k0 = 0, it = 0; k0 < DIN; k0 += 32, ++it) {
    const int cur = it & 1;
    if (k0 + 32 < DIN) stage_all(cur ^ 1, k0 + 32);   // overlap DMA with WMMA

    // A fragment (16x32 bf16, ISA 7.12.2): lanes 0-15 hold K 0..7 & 16..23 of
    // row mw+l; lanes 16-31 hold K 8..15 & 24..31.
    Frag32 ah, al;
    {
      const unsigned short* p = &sAhi[cur][mw + l16][top ? 8 : 0];
      ah.q[0] = *(const uint4*)p;
      ah.q[1] = *(const uint4*)(p + 16);
      const unsigned short* q = &sAlo[cur][mw + l16][top ? 8 : 0];
      al.q[0] = *(const uint4*)q;
      al.q[1] = *(const uint4*)(q + 16);
    }

#pragma unroll
    for (int tb = 0; tb < NB; ++tb) {
#pragma unroll
      for (int s = 0; s < 2; ++s) {
        const int nb = nw + s * 16;
        // B fragment (32x16): lanes 0-15 hold K 0..7 & 16..23 of col nb+l,
        // lanes 16-31 hold K 8..15 & 24..31 (same order as A).
        Frag32 bh, bl;
        const unsigned short* p = &sBhi[cur][tb][nb + l16][top ? 8 : 0];
        bh.q[0] = *(const uint4*)p;
        bh.q[1] = *(const uint4*)(p + 16);
        const unsigned short* q = &sBlo[cur][tb][nb + l16][top ? 8 : 0];
        bl.q[0] = *(const uint4*)q;
        bl.q[1] = *(const uint4*)(q + 16);

        v8f c = acc[tb][s];
        c = __builtin_amdgcn_wmma_f32_16x16x32_bf16(false, ah.v, false, bh.v, (short)0, c, false, false);
        c = __builtin_amdgcn_wmma_f32_16x16x32_bf16(false, al.v, false, bh.v, (short)0, c, false, false);
        c = __builtin_amdgcn_wmma_f32_16x16x32_bf16(false, ah.v, false, bl.v, (short)0, c, false, false);
        acc[tb][s] = c;
      }
    }

    wait_async0();        // next buffer landed
    __syncthreads();      // everyone done reading cur, next tile visible
  }

  // Epilogue. C/D layout: VGPR i -> (M = i (+8 for lanes>=16), N = lane%16).
#pragma unroll
  for (int s = 0; s < 2; ++s) {
    const int col = n0 + nw + s * 16 + l16;
#pragma unroll
    for (int i = 0; i < 8; ++i) {
      const long row = m0 + mw + i + (top ? 8 : 0);
      if (MODE == 0) {
        const float hd = acc[0][s][i];
        const float g  = acc[NB - 1][s][i];
        const float la = -softplusf(g);                                    // log_coeffs
        const float lg = (hd >= 0.0f) ? logf(hd + 0.5f) : -softplusf(-hd); // log_g
        const float lv = -softplusf(-g) + lg;                              // log_values
        out0[row * HID + col] = la;
        out1[row * HID + col] = lv;
      } else {
        out0[row * DOUT + col] = acc[0][s][i];
      }
    }
  }
}

// ---------- chunked log-space scan (3 passes) ----------
// gid mapping for pass 1/3: h = gid&1023, chunk c = (gid>>10)&15, b = gid>>14.

__global__ __launch_bounds__(256)
void scan_pass1(const float* __restrict__ Aarr, const float* __restrict__ Varr,
                const unsigned char* __restrict__ init,
                float* __restrict__ ca, float* __restrict__ cb,
                unsigned char* __restrict__ cr) {
  const int gid = blockIdx.x * 256 + threadIdx.x;
  const int h = gid & (HID - 1);
  const int c = (gid >> 10) & (NCHUNK - 1);
  const int b = gid >> 14;
  float accA = 0.0f, accB = -INFINITY;
  unsigned char accR = 0;
  const long tbase = (long)b * TSEQ + c * CLEN;
  for (int j = 0; j < CLEN; ++j) {
    const long t   = tbase + j;
    const long idx = t * HID + h;
    const float a = Aarr[idx];
    const float v = Varr[idx];
    const bool  r = init[t] != 0;
    const float bin = r ? -INFINITY : accB;
    accB = logaddexpf_(a + bin, v);
    accA = a + (r ? 0.0f : accA);
    accR |= (unsigned char)r;
  }
  ca[gid] = accA;
  cb[gid] = accB;
  cr[gid] = accR;
}

__global__ __launch_bounds__(256)
void scan_pass2(const float* __restrict__ ca, const float* __restrict__ cb,
                const unsigned char* __restrict__ cr, float* __restrict__ exB) {
  const int gid = blockIdx.x * 256 + threadIdx.x;   // one thread per (b,h)
  const int h = gid & (HID - 1);
  const int b = gid >> 10;
  float Bv = -INFINITY;                              // exclusive prefix state
  for (int c = 0; c < NCHUNK; ++c) {
    const int i = (b * NCHUNK + c) * HID + h;
    exB[i] = Bv;
    const float bin = cr[i] ? -INFINITY : Bv;
    Bv = logaddexpf_(ca[i] + bin, cb[i]);
  }
}

// Pass 3 emits h directly as bf16 hi/lo planes (GEMM2 operands) + f32 h_n.
__global__ __launch_bounds__(256)
void scan_pass3(const float* __restrict__ Aarr, const float* __restrict__ Varr,
                const unsigned char* __restrict__ init,
                const float* __restrict__ exB,
                unsigned short* __restrict__ Hhi, unsigned short* __restrict__ Hlo,
                float* __restrict__ hn) {
  const int gid = blockIdx.x * 256 + threadIdx.x;
  const int h = gid & (HID - 1);
  const int c = (gid >> 10) & (NCHUNK - 1);
  const int b = gid >> 14;
  float logB = exB[gid];
  const long tbase = (long)b * TSEQ + c * CLEN;
  float hv = 0.0f;
  for (int j = 0; j < CLEN; ++j) {
    const long t   = tbase + j;
    const long idx = t * HID + h;
    const float a = Aarr[idx];
    const float v = Varr[idx];
    const float bin = (init[t] != 0) ? -INFINITY : logB;
    logB = logaddexpf_(a + bin, v);
    hv = expf(logB);
    unsigned short hh, hl;
    split_hi_lo(hv, &hh, &hl);
    Hhi[idx] = hh;
    Hlo[idx] = hl;
  }
  if (c == NCHUNK - 1) hn[b * HID + h] = hv;
}

// ---------- launcher ----------

extern "C" void kernel_launch(void* const* d_in, const int* in_sizes, int n_in,
                              void* d_out, int out_size, void* d_ws, size_t ws_size,
                              hipStream_t stream) {
  const float*         x       = (const float*)d_in[0];
  const unsigned char* is_init = (const unsigned char*)d_in[1];  // jnp bool: 1 byte
  const float*         Whg     = (const float*)d_in[2];
  const float*         Wout    = (const float*)d_in[3];

  float* out = (float*)d_out;                       // (8,4096,1024) f32
  float* hn  = out + (long)TOTAL_M * DOUT;          // (8,1,1024) f32

  // workspace (~410MB): bf16 planes + f32 a/v + chunk summaries.
  unsigned short* xhi  = (unsigned short*)d_ws;                 // [M][K]
  unsigned short* xlo  = xhi + (long)TOTAL_M * DIN;
  unsigned short* whgT_hi  = xlo + (long)TOTAL_M * DIN;         // [2048][1024]
  unsigned short* whgT_lo  = whgT_hi + (long)2 * HID * DIN;
  unsigned short* woutT_hi = whgT_lo + (long)2 * HID * DIN;     // [1024][1024]
  unsigned short* woutT_lo = woutT_hi + (long)DOUT * DIN;
  float* wsA  = (float*)(woutT_lo + (long)DOUT * DIN);          // log_coeffs
  float* wsV  = wsA + (long)TOTAL_M * HID;                      // log_values
  float* wsCA = wsV + (long)TOTAL_M * HID;
  float* wsCB = wsCA + BATCH * HID * NCHUNK;
  float* wsEx = wsCB + BATCH * HID * NCHUNK;
  unsigned char* wsCR = (unsigned char*)(wsEx + BATCH * HID * NCHUNK);
  // h bf16 planes alias the x planes (x is dead after GEMM1).
  unsigned short* Hhi = xhi;
  unsigned short* Hlo = xlo;

  // 1) one-time conversions (also halves GEMM global traffic)
  convert_x <<<(long)TOTAL_M * DIN / 8 / 256, 256, 0, stream>>>(x, xhi, xlo);
  convert_wT<<<(2 * HID) * (DIN / 8) / 256,   256, 0, stream>>>(Whg, 2 * HID, whgT_hi, whgT_lo);
  convert_wT<<<DOUT * (DIN / 8) / 256,        256, 0, stream>>>(Wout, DOUT, woutT_hi, woutT_lo);

  // 2) GEMM1 (hidden+gate fused epilogue -> log_coeffs/log_values)
  dim3 g1(HID / 64, TOTAL_M / 64);
  wmma_gemm<0><<<g1, 256, 0, stream>>>(xhi, xlo, whgT_hi, whgT_lo, wsA, wsV);

  // 3) chunked scan
  scan_pass1<<<(BATCH * HID * NCHUNK) / 256, 256, 0, stream>>>(wsA, wsV, is_init, wsCA, wsCB, wsCR);
  scan_pass2<<<(BATCH * HID) / 256,          256, 0, stream>>>(wsCA, wsCB, wsCR, wsEx);
  scan_pass3<<<(BATCH * HID * NCHUNK) / 256, 256, 0, stream>>>(wsA, wsV, is_init, wsEx, Hhi, Hlo, hn);

  // 4) GEMM2: out = h @ W_out
  dim3 g2(DOUT / 64, TOTAL_M / 64);
  wmma_gemm<1><<<g2, 256, 0, stream>>>(Hhi, Hlo, woutT_hi, woutT_lo, out, nullptr);
}
